// QNetwork_45389214384165
// MI455X (gfx1250) — matrix operand order repaired and checked by reading, not compiled, not run on hardware
//
#include <hip/hip_runtime.h>
#include <hip/hip_bf16.h>

// ---------------------------------------------------------------------------
// GRU Q-network for MI455X (gfx1250, wave32, WMMA bf16).
//   T=4096 B=64 F=64 H=128 3H=384 A=16
// Phase 1 (prep):   transpose+convert weights to bf16 (W^T row-major so WMMA
//                   B-fragments are contiguous loads).
// Phase 2 (enc):    xi = relu(x@W0+b0)@Wi, tiled WMMA, stored bf16 in ws.
// Phase 3 (scan):   4 workgroups (16 batch rows each), Wh fragments held in
//                   VGPRs across all 4096 steps, q-projection fused, per-step
//                   xi slice double-buffered in LDS via async global->LDS
//                   copies (ASYNCcnt) overlapped with the sequential chain.
// ---------------------------------------------------------------------------

#define TT 4096
#define BB 64
#define FF 64
#define HH 128
#define H3 384
#define AA 16

typedef __attribute__((ext_vector_type(16))) __bf16 v16bf;
typedef __attribute__((ext_vector_type(8)))  float  v8f;
typedef int v4i_ __attribute__((vector_size(16)));   // 128-bit payload type

union FragU {
    v16bf v;
    uint4 q[2];
    unsigned short s[16];
};

__device__ __forceinline__ unsigned short f2bf(float f) {
    unsigned int u = __float_as_uint(f);
    unsigned int r = u + 0x7FFFu + ((u >> 16) & 1u);   // round-to-nearest-even
    return (unsigned short)(r >> 16);
}
__device__ __forceinline__ float bf2f(unsigned short s) {
    return __uint_as_float(((unsigned int)s) << 16);
}

// A-fragment (16x32 bf16, MxK): lane<16 -> row m0+lane, K in [k0..k0+15]
//                               lane>=16 -> row, K in [k0+16..k0+31]
__device__ __forceinline__ v16bf ldA(const unsigned short* base, int ld, int m0,
                                     int k0, int lane) {
    int r  = m0 + (lane & 15);
    int kk = k0 + ((lane >> 4) << 4);
    const unsigned short* p = base + r * ld + kk;
    FragU f;
    f.q[0] = *(const uint4*)(p);
    f.q[1] = *(const uint4*)(p + 8);
    return f.v;
}

// B-fragment (32x16 bf16, KxN) from transposed weights WT[N][ldk] (row-major):
// lane<16 -> col n0+lane, K in [k0..k0+15]; lane>=16 -> col, K in [k0+16..k0+31]
__device__ __forceinline__ v16bf ldB(const unsigned short* WT, int ldk, int n0,
                                     int k0, int lane) {
    int n  = n0 + (lane & 15);
    int kk = k0 + ((lane >> 4) << 4);
    const unsigned short* p = WT + n * ldk + kk;
    FragU f;
    f.q[0] = *(const uint4*)(p);
    f.q[1] = *(const uint4*)(p + 8);
    return f.v;
}

__device__ __forceinline__ v8f wmma_bf16(v16bf a, v16bf b, v8f c) {
    return __builtin_amdgcn_wmma_f32_16x16x32_bf16(false, a, false, b,
                                                   (short)0, c, false, false);
}

// --- async global->LDS staging (gfx1250 ASYNCcnt path, guarded) -------------
#if defined(__gfx1250__) && __has_builtin(__builtin_amdgcn_global_load_async_to_lds_b128)
#define XI_ASYNC 1
#else
#define XI_ASYNC 0
#endif

__device__ __forceinline__ void wait_async0() {
#if XI_ASYNC
#if __has_builtin(__builtin_amdgcn_s_wait_asynccnt)
    __builtin_amdgcn_s_wait_asynccnt(0);
#else
    asm volatile("s_wait_asynccnt 0" ::: "memory");
#endif
#endif
}

// copy one step's xi slice: 16 rows x 384 bf16 = 12288 B, contiguous.
// 768 x b128 transfers, 3 per thread (256 threads).
__device__ __forceinline__ void stage_xi(const unsigned short* __restrict__ src,
                                         unsigned short* dstLds, int tid) {
#if XI_ASYNC
    for (int e = 0; e < 3; ++e) {
        int off = (tid + e * 256) * 8;
        __builtin_amdgcn_global_load_async_to_lds_b128(
            (__attribute__((address_space(1))) v4i_*)(src + off),
            (__attribute__((address_space(3))) v4i_*)(dstLds + off),
            0, 0);
    }
#else
    for (int e = 0; e < 3; ++e) {
        int off = (tid + e * 256) * 8;
        *(uint4*)(dstLds + off) = *(const uint4*)(src + off);
    }
#endif
}

// ---------------------------------------------------------------------------
// Phase 1: weight transpose + f32 -> bf16
// ---------------------------------------------------------------------------
__global__ void prep_kernel(const float* __restrict__ W0, const float* __restrict__ Wi,
                            const float* __restrict__ Wh, const float* __restrict__ Wq,
                            unsigned short* __restrict__ W0T, unsigned short* __restrict__ WiT,
                            unsigned short* __restrict__ WhT, unsigned short* __restrict__ WqT) {
    int i = blockIdx.x * blockDim.x + threadIdx.x;
    if (i < FF * HH) {                 // W0[F][H] -> W0T[H][F]
        int k = i / HH, n = i % HH;
        W0T[n * FF + k] = f2bf(W0[i]);
    }
    if (i < HH * H3) {                 // Wi/Wh[H][3H] -> [3H][H]
        int k = i / H3, n = i % H3;
        WiT[n * HH + k] = f2bf(Wi[i]);
        WhT[n * HH + k] = f2bf(Wh[i]);
    }
    if (i < HH * AA) {                 // Wq[H][A] -> WqT[A][H]
        int k = i / AA, n = i % AA;
        WqT[n * HH + k] = f2bf(Wq[i]);
    }
}

// ---------------------------------------------------------------------------
// Phase 2: xi = relu(x@W0 + b0) @ Wi   (rows = T*B flattened, 64-row slabs)
// 4 waves: wave w owns M-tile w; N looped (8 tiles for feats, 24 for xi).
// ---------------------------------------------------------------------------
__global__ __launch_bounds__(128) void enc_kernel(
    const float* __restrict__ x, const float* __restrict__ b0v,
    const unsigned short* __restrict__ W0T, const unsigned short* __restrict__ WiT,
    unsigned short* __restrict__ xi) {
    __shared__ __align__(16) unsigned short Xs[64 * 72];    // bf16 x slab, pad+8
    __shared__ __align__(16) unsigned short Fs[64 * 136];   // bf16 feats, pad+8

    const int tid = threadIdx.x, lane = tid & 31, wave = tid >> 5;
    const long g0 = (long)blockIdx.x * 64;

    // stage x slab -> bf16 LDS
    for (int e = 0; e < 32; ++e) {
        int f = tid + e * 128;
        int r = f >> 6, c = f & 63;
        Xs[r * 72 + c] = f2bf(x[(g0 + r) * FF + c]);
    }
    __syncthreads();

    const int m0 = wave * 16;
    const int mb = m0 + ((lane >> 4) << 3);   // C-fragment row base

    // GEMM1: feats = relu(X @ W0 + b0)   M=16 N=128 K=64
    v8f acc[8];
    for (int j = 0; j < 8; ++j) acc[j] = (v8f){};
    for (int k0 = 0; k0 < FF; k0 += 32) {
        v16bf a = ldA(Xs, 72, m0, k0, lane);
        for (int j = 0; j < 8; ++j) {
            v16bf b = ldB(W0T, FF, j * 16, k0, lane);
            acc[j] = wmma_bf16(a, b, acc[j]);
        }
    }
    for (int j = 0; j < 8; ++j) {
        int n = j * 16 + (lane & 15);
        float bias = b0v[n];
        for (int i = 0; i < 8; ++i) {
            float v = fmaxf(acc[j][i] + bias, 0.0f);
            Fs[(mb + i) * 136 + n] = f2bf(v);
        }
    }
    __syncthreads();

    // GEMM2: xi = feats @ Wi   M=16 N=384 K=128, streamed over N
    v16bf af[4];
    for (int k = 0; k < 4; ++k) af[k] = ldA(Fs, 136, m0, k * 32, lane);
    for (int j = 0; j < 24; ++j) {
        v8f c = (v8f){};
        for (int k = 0; k < 4; ++k) {
            v16bf b = ldB(WiT, HH, j * 16, k * 32, lane);
            c = wmma_bf16(af[k], b, c);
        }
        int n = j * 16 + (lane & 15);
        for (int i = 0; i < 8; ++i)
            xi[(g0 + mb + i) * H3 + n] = f2bf(c[i]);
    }
}

// ---------------------------------------------------------------------------
// Phase 3: sequential GRU scan + fused q projection.
// grid = 4 workgroups (16 batch rows each), 8 waves; Wh^T fragments live in
// VGPRs for the whole 4096-step loop (3 N-tiles x 4 K-tiles per wave).
// Per-step xi slice is double-buffered in LDS via async copies.
// ---------------------------------------------------------------------------
__global__ __launch_bounds__(256) void scan_kernel(
    const unsigned char* __restrict__ mask, const float* __restrict__ bh,
    const float* __restrict__ bq, const float* __restrict__ ic,
    const unsigned short* __restrict__ WhT, const unsigned short* __restrict__ WqT,
    const unsigned short* __restrict__ xi,
    float* __restrict__ out_h, float* __restrict__ out_q) {
    __shared__ __align__(16) unsigned short hb[16 * 136];  // masked h (bf16, A-side)
    __shared__ __align__(16) float          hf[16 * 136];  // masked h (f32)
    __shared__ __align__(16) unsigned short hq[16 * 136];  // unmasked h_new (bf16, for q)
    __shared__ __align__(16) float          ghs[16 * 392]; // gate pre-activations
    __shared__ __align__(16) float          hnf[16 * 128]; // unmasked h_new (f32, final)
    __shared__ __align__(16) unsigned short xis[2][16 * 384];  // double-buffered xi slice

    const int tid = threadIdx.x, lane = tid & 31, wave = tid >> 5;
    const int b0 = blockIdx.x * 16;
    const int mb = (lane >> 4) << 3;

    // Register-resident Wh B-fragments: wave owns N-tiles {3w, 3w+1, 3w+2}
    FragU Bf[3][4];
    float bhv[3];
    for (int jj = 0; jj < 3; ++jj) {
        int j = wave * 3 + jj;
        int n = j * 16 + (lane & 15);
        bhv[jj] = bh[n];
        for (int k = 0; k < 4; ++k) {
            const unsigned short* p = WhT + n * HH + k * 32 + ((lane >> 4) << 4);
            Bf[jj][k].q[0] = *(const uint4*)(p);
            Bf[jj][k].q[1] = *(const uint4*)(p + 8);
        }
    }
    // Wq fragments (used by wave 0, cheap to hold everywhere)
    FragU Bq[4];
    float bqv = bq[lane & 15];
    for (int k = 0; k < 4; ++k) {
        const unsigned short* p = WqT + (lane & 15) * HH + k * 32 + ((lane >> 4) << 4);
        Bq[k].q[0] = *(const uint4*)(p);
        Bq[k].q[1] = *(const uint4*)(p + 8);
    }

    // h = initial_carry (mask at t=0 selects ic either way)
    for (int e = 0; e < 8; ++e) {
        int f = tid + e * 256;
        int m = f >> 7, n = f & 127;
        float v = ic[(b0 + m) * HH + n];
        hf[m * 136 + n] = v;
        hb[m * 136 + n] = f2bf(v);
        hnf[m * 128 + n] = v;
    }
    // kick off async stage of xi slice for t=0
    stage_xi(xi + (long)b0 * H3, &xis[0][0], tid);

    int cur = 0;
    for (int t = 0; t < TT; ++t) {
        const long rowt = (long)t * BB + b0;

        wait_async0();           // xi slice for step t landed in xis[cur]
        __syncthreads();         // ..and is visible to all waves (also fences hb/hq)
        if (t + 1 < TT)          // overlap next slice with this step's compute
            stage_xi(xi + (rowt + BB) * H3, &xis[cur ^ 1][0], tid);
        const unsigned short* xs = &xis[cur][0];

        // gh = h @ Wh ; stash gh + bh (+ xi for r/z chunks) in LDS
        v16bf afr[4];
        for (int k = 0; k < 4; ++k) afr[k] = ldA(hb, 136, 0, k * 32, lane);
        for (int jj = 0; jj < 3; ++jj) {
            int j = wave * 3 + jj;
            int n = j * 16 + (lane & 15);
            v8f acc = (v8f){};
            for (int k = 0; k < 4; ++k)
                acc = wmma_bf16(afr[k], Bf[jj][k].v, acc);
            for (int i = 0; i < 8; ++i) {
                float v = acc[i] + bhv[jj];
                if (j < 16) v += bf2f(xs[(mb + i) * H3 + n]);  // xr/xz folded in
                ghs[(mb + i) * 392 + n] = v;
            }
        }
        __syncthreads();

        // gates + carry update (mask for t+1 folded in; unmasked h_new kept for q)
        for (int e = 0; e < 8; ++e) {
            int f = tid + e * 256;
            int m = f >> 7, n = f & 127;
            float pre_r = ghs[m * 392 + n];
            float pre_z = ghs[m * 392 + 128 + n];
            float hn    = ghs[m * 392 + 256 + n];
            float xn    = bf2f(xs[m * H3 + 256 + n]);
            float r = 1.0f / (1.0f + __expf(-pre_r));
            float z = 1.0f / (1.0f + __expf(-pre_z));
            float nn = tanhf(xn + r * hn);
            float hnew = (1.0f - z) * nn + z * hf[m * 136 + n];
            hnf[m * 128 + n] = hnew;
            hq[m * 136 + n]  = f2bf(hnew);
            float hm = hnew;
            if (t + 1 < TT && mask[(t + 1) * BB + b0 + m]) hm = ic[(b0 + m) * HH + n];
            hf[m * 136 + n] = hm;
            hb[m * 136 + n] = f2bf(hm);
        }
        __syncthreads();

        // q_t = h_new @ Wq + bq  (one 16x16 tile; wave 0)
        if (wave == 0) {
            v8f qa = (v8f){};
            for (int k = 0; k < 4; ++k) {
                v16bf aq = ldA(hq, 136, 0, k * 32, lane);
                qa = wmma_bf16(aq, Bq[k].v, qa);
            }
            for (int i = 0; i < 8; ++i)
                out_q[(rowt + mb + i) * AA + (lane & 15)] = qa[i] + bqv;
        }
        // no barrier needed here: next iteration's top barrier protects hq/hb/xis
        cur ^= 1;
    }

    for (int e = 0; e < 8; ++e) {
        int f = tid + e * 256;
        int m = f >> 7, n = f & 127;
        out_h[(b0 + m) * HH + n] = hnf[m * 128 + n];
    }
}

// ---------------------------------------------------------------------------
extern "C" void kernel_launch(void* const* d_in, const int* in_sizes, int n_in,
                              void* d_out, int out_size, void* d_ws, size_t ws_size,
                              hipStream_t stream) {
    const float*         x    = (const float*)d_in[0];
    const unsigned char* mask = (const unsigned char*)d_in[1];  // jax bool = 1 byte
    const float*         W0   = (const float*)d_in[2];
    const float*         b0   = (const float*)d_in[3];
    const float*         Wi   = (const float*)d_in[4];
    const float*         Wh   = (const float*)d_in[5];
    const float*         bh   = (const float*)d_in[6];
    const float*         Wq   = (const float*)d_in[7];
    const float*         bq   = (const float*)d_in[8];
    const float*         ic   = (const float*)d_in[9];

    char* ws = (char*)d_ws;
    unsigned short* W0T = (unsigned short*)(ws);                       //  16 KB
    unsigned short* WiT = (unsigned short*)(ws + 16384);               //  96 KB
    unsigned short* WhT = (unsigned short*)(ws + 16384 + 98304);       //  96 KB
    unsigned short* WqT = (unsigned short*)(ws + 16384 + 2 * 98304);   //   4 KB
    unsigned short* xi  = (unsigned short*)(ws + 16384 + 2 * 98304 + 4096);  // 192 MB

    float* out_h = (float*)d_out;            // [B,H] = 8192 floats
    float* out_q = out_h + BB * HH;          // [T,B,A]

    prep_kernel<<<(HH * H3 + 255) / 256, 256, 0, stream>>>(W0, Wi, Wh, Wq,
                                                           W0T, WiT, WhT, WqT);
    enc_kernel<<<(TT * BB) / 64, 128, 0, stream>>>(x, b0, W0T, WiT, xi);
    scan_kernel<<<BB / 16, 256, 0, stream>>>(mask, bh, bq, ic, WhT, WqT, xi,
                                             out_h, out_q);
}
